// CompatibilityGraphEmbedding_77214922047592
// MI455X (gfx1250) — compile-verified, matrix operand order checked by reading, not run on hardware
//
#include <hip/hip_runtime.h>
#include <hip/hip_bf16.h>

// ---------------------------------------------------------------------------
// CompatibilityGraphEmbedding on MI455X (gfx1250, wave32, WMMA 16x16x32 f16)
// ---------------------------------------------------------------------------

typedef __attribute__((ext_vector_type(16))) _Float16 v16h;
typedef __attribute__((ext_vector_type(8)))  _Float16 v8h;
typedef __attribute__((ext_vector_type(8)))  float    v8f;

constexpr int kN  = 6144;
constexpr int kC  = 128;
constexpr int kBN = 64;                 // attention column-block
constexpr int kNB = kN / kBN;           // 96 column blocks
constexpr int kSP = 4;                  // attention column splits (split-K flash)
constexpr int kNBS = kNB / kSP;         // 24 column blocks per split

// ---- shared fragment loader: 16x32 f16 tile of a row-major matrix ---------
// A-matrix layout (ISA 7.12.2): lane L (L<16) = row L, elems {K 0..7, 16..23};
// lane L+16 = row L, elems {K 8..15, 24..31}.  B mirrors this per column.
__device__ __forceinline__ v16h load_frag(const _Float16* p0, int ld) {
  int lane = threadIdx.x & 31;
  int r    = lane & 15;
  int koff = (lane >> 4) << 3;          // 0 or 8
  const _Float16* p = p0 + r * ld + koff;
  v8h lo = *(const v8h*)(p);
  v8h hi = *(const v8h*)(p + 16);
  v16h f;
#pragma unroll
  for (int i = 0; i < 8; ++i) { f[i] = lo[i]; f[i + 8] = hi[i]; }
  return f;
}

__device__ __forceinline__ v8f wmma_f16(v16h a, v16h b, v8f c) {
  return __builtin_amdgcn_wmma_f32_16x16x32_f16(false, a, false, b,
                                                (short)0, c, false, false);
}

// ---------------------------------------------------------------------------
// weight convert+transpose: W[mats][128][128] f32 (k-major) -> Wt[mats][n][k] f16
// ---------------------------------------------------------------------------
__global__ void tconv_kernel(const float* __restrict__ W,
                             _Float16* __restrict__ Wt, int total) {
  int idx = blockIdx.x * blockDim.x + threadIdx.x;
  if (idx >= total) return;
  int m = idx >> 14;                 // / 16384
  int r = idx & 16383;
  int k = r >> 7, n = r & 127;
  Wt[m * 16384 + n * 128 + k] = (_Float16)W[idx];
}

// ---------------------------------------------------------------------------
// embedding: feat = [corr_feat | centered kp] @ W_in + b_in
// ---------------------------------------------------------------------------
__global__ void embed_kernel(const float* __restrict__ refk,
                             const float* __restrict__ srck,
                             const float* __restrict__ cf,
                             const float* __restrict__ Win,
                             const float* __restrict__ bin,
                             float* __restrict__ feat,
                             _Float16* __restrict__ featH) {
  int idx = blockIdx.x * blockDim.x + threadIdx.x;
  if (idx >= kN * kC) return;
  int n = idx >> 7, c = idx & 127;
  float kp[6];
  kp[0] = refk[n * 3 + 0]; kp[1] = refk[n * 3 + 1]; kp[2] = refk[n * 3 + 2];
  kp[3] = srck[n * 3 + 0]; kp[4] = srck[n * 3 + 1]; kp[5] = srck[n * 3 + 2];
  float m = (kp[0] + kp[1] + kp[2] + kp[3] + kp[4] + kp[5]) * (1.0f / 6.0f);
  float acc = bin[c];
#pragma unroll
  for (int k = 0; k < 6; ++k) acc += cf[n * 6 + k] * Win[k * 128 + c];
#pragma unroll
  for (int k = 0; k < 6; ++k) acc += (kp[k] - m) * Win[(6 + k) * 128 + c];
  feat[idx] = acc;
  featH[idx] = (_Float16)acc;
}

// ---------------------------------------------------------------------------
// GEMM: Out[N,128] = A[N,128](f16) @ Wt^T + bias.   Wt is f16 [128 out][128 in].
// Block: 256 thr = 8 waves; block covers 16 rows x 128 cols (wave = 16x16 tile).
// ---------------------------------------------------------------------------
template <int STOREHALF>
__global__ void gemm_nt_kernel(const _Float16* __restrict__ A,
                               const _Float16* __restrict__ Bt,
                               const float* __restrict__ bias,
                               void* __restrict__ Out) {
  __shared__ _Float16 As[16 * kC];
  int t = threadIdx.x;
  int rowbase = blockIdx.x * 16;
  { // stage A tile: 2048 halves / 256 threads
    const v8h* src = (const v8h*)(A + rowbase * kC);
    ((v8h*)As)[t] = src[t];
  }
  __syncthreads();
  int wave = t >> 5, lane = t & 31;
  int nbase = wave * 16;
  v8f acc = {};
#pragma unroll
  for (int kc = 0; kc < 4; ++kc) {
    v16h a = load_frag(As + kc * 32, kC);
    v16h b = load_frag(Bt + nbase * kC + kc * 32, kC);
    acc = wmma_f16(a, b, acc);
  }
  int col  = nbase + (lane & 15);
  int roff = (lane >> 4) * 8;
  float bv = bias[col];
#pragma unroll
  for (int j = 0; j < 8; ++j) {
    int row = rowbase + j + roff;
    float v = acc[j] + bv;
    if (STOREHALF) ((_Float16*)Out)[row * kC + col] = (_Float16)v;
    else           ((float*)Out)[row * kC + col] = v;
  }
}

// ---------------------------------------------------------------------------
// LayerNorm (+optional residual, +optional LeakyReLU 0.1); one row per block.
// ---------------------------------------------------------------------------
__global__ void ln_kernel(const float* __restrict__ X,
                          const float* __restrict__ Res,
                          const float* __restrict__ g,
                          const float* __restrict__ b,
                          float* __restrict__ OutF,
                          _Float16* __restrict__ OutH, int leaky) {
  int row = blockIdx.x, t = threadIdx.x;
  float x = X[row * kC + t];
  if (Res) x += Res[row * kC + t];
  __shared__ float red0[4], red1[4];
  float s = x, s2 = x * x;
#pragma unroll
  for (int m = 16; m >= 1; m >>= 1) {
    s  += __shfl_xor(s, m, 32);
    s2 += __shfl_xor(s2, m, 32);
  }
  int lane = t & 31, wv = t >> 5;
  if (lane == 0) { red0[wv] = s; red1[wv] = s2; }
  __syncthreads();
  s  = red0[0] + red0[1] + red0[2] + red0[3];
  s2 = red1[0] + red1[1] + red1[2] + red1[3];
  float mean = s * (1.0f / 128.0f);
  float var  = s2 * (1.0f / 128.0f) - mean * mean;
  float y = (x - mean) * rsqrtf(var + 1e-5f) * g[t] + b[t];
  if (leaky) y = (y > 0.0f) ? y : 0.1f * y;
  OutF[row * kC + t] = y;
  OutH[row * kC + t] = (_Float16)y;
}

// ---------------------------------------------------------------------------
// Split-K flash attention with multiplicative geo mask recomputed from
// keypoints.  Grid: (N/64 row-blocks, kSP column-splits); 128 thr = 4 waves;
// each wave owns 16 query rows.  Row keypoint data lives in LDS (2 broadcast
// ds_load_b128 per j) to keep per-lane VGPR pressure below the spill point.
// ---------------------------------------------------------------------------
__global__ void __launch_bounds__(128, 1)
attn_kernel(const _Float16* __restrict__ Q,
            const _Float16* __restrict__ K,
            const _Float16* __restrict__ V,
            const float* __restrict__ refk,
            const float* __restrict__ srck,
            float* __restrict__ Opart,   // [kSP][N][C]
            float* __restrict__ mstat,   // [kSP][N]
            float* __restrict__ lstat) { // [kSP][N]
  __shared__ _Float16 Ks[kBN * kC];       // [n][c] row-major
  __shared__ _Float16 Vt[kC * kBN];       // [c][n] (transposed V)
  __shared__ float    ckp[6][kBN];        // column keypoints
  __shared__ float    cn2[2][kBN];        // column |.|^2
  __shared__ float    rowdat[64 * 8];     // per-row {rx,ry,rz,rn2,sx,sy,sz,sn2}
  __shared__ _Float16 Ps[4][16 * kBN];    // per-wave P tile (A-fragment layout)

  int t = threadIdx.x;
  int wave = t >> 5, lane = t & 31;
  int mbase = blockIdx.x * 64 + wave * 16;
  int split = blockIdx.y;
  int roff  = (lane >> 4) * 8;            // C/D-fragment row offset per half
  int nl    = lane & 15;
  int w16   = wave * 16;

  // stage the block's 64 rows of keypoint data into LDS (once)
  if (t < 64) {
    int m = blockIdx.x * 64 + t;
    float ax = refk[m*3], ay = refk[m*3+1], az = refk[m*3+2];
    float bx = srck[m*3], by = srck[m*3+1], bz = srck[m*3+2];
    float* p = rowdat + t * 8;
    p[0] = ax; p[1] = ay; p[2] = az; p[3] = ax*ax + ay*ay + az*az;
    p[4] = bx; p[5] = by; p[6] = bz; p[7] = bx*bx + by*by + bz*bz;
  }

  v16h qf[4];
#pragma unroll
  for (int kc = 0; kc < 4; ++kc) qf[kc] = load_frag(Q + mbase * kC + kc * 32, kC);

  v8f o[8];
  v8f zero = {};
#pragma unroll
  for (int i = 0; i < 8; ++i) o[i] = zero;
  float mrun[8], ls[8];
#pragma unroll
  for (int j = 0; j < 8; ++j) { mrun[j] = -1e30f; ls[j] = 0.0f; }

  const float SCALE = 0.08838834764831845f;  // 1/sqrt(128)
  const float INVS2 = 1.0f / (0.3f * 0.3f);

  for (int nb = split * kNBS; nb < (split + 1) * kNBS; ++nb) {
    int nbase = nb * kBN;
    __syncthreads();
    { // stage K: 8192 halves, 128 threads
      const v8h* src = (const v8h*)(K + nbase * kC);
      v8h* dst = (v8h*)Ks;
#pragma unroll
      for (int i = 0; i < 8; ++i) dst[t + i * 128] = src[t + i * 128];
    }
    { // stage V transposed
#pragma unroll
      for (int i = 0; i < 8; ++i) {
        int base = (t + i * 128) * 8;
        v8h vv = *(const v8h*)(V + nbase * kC + base);
        int n = base >> 7, c0 = base & 127;
#pragma unroll
        for (int e = 0; e < 8; ++e) Vt[(c0 + e) * kBN + n] = vv[e];
      }
    }
    if (t < kBN) { // stage column keypoints
      int n = nbase + t;
      float a0 = refk[n*3], a1 = refk[n*3+1], a2 = refk[n*3+2];
      float b0 = srck[n*3], b1 = srck[n*3+1], b2 = srck[n*3+2];
      ckp[0][t] = a0; ckp[1][t] = a1; ckp[2][t] = a2;
      ckp[3][t] = b0; ckp[4][t] = b1; ckp[5][t] = b2;
      cn2[0][t] = a0*a0 + a1*a1 + a2*a2;
      cn2[1][t] = b0*b0 + b1*b1 + b2*b2;
    }
    __syncthreads();

    // --- raw scores S = q k^T for 4 16-col subtiles (WMMA) ----------------
    float sv[4][8];
#pragma unroll
    for (int st = 0; st < 4; ++st) {
      v8f s = {};
#pragma unroll
      for (int kc = 0; kc < 4; ++kc) {
        v16h b = load_frag(Ks + (st * 16) * kC + kc * 32, kC);
        s = wmma_f16(qf[kc], b, s);
      }
#pragma unroll
      for (int j = 0; j < 8; ++j) sv[st][j] = s[j];
    }

    // --- geo mask: column values in transient regs, row values from LDS ---
    float ccx[4], ccy[4], ccz[4], cux[4], cuy[4], cuz[4], cc2r[4], cc2s[4];
#pragma unroll
    for (int st = 0; st < 4; ++st) {
      int n = st * 16 + nl;
      ccx[st] = ckp[0][n]; ccy[st] = ckp[1][n]; ccz[st] = ckp[2][n];
      cux[st] = ckp[3][n]; cuy[st] = ckp[4][n]; cuz[st] = ckp[5][n];
      cc2r[st] = cn2[0][n]; cc2s[st] = cn2[1][n];
    }
    float mnew[8];
#pragma unroll
    for (int j = 0; j < 8; ++j) {
      const float* rp = rowdat + (w16 + j + roff) * 8;
      float4 r0 = *(const float4*)(rp);       // rx, ry, rz, rn2
      float4 r1 = *(const float4*)(rp + 4);   // sx, sy, sz, sn2
      mnew[j] = mrun[j];
#pragma unroll
      for (int st = 0; st < 4; ++st) {
        float d2r = r0.w + cc2r[st] - 2.0f * (r0.x*ccx[st] + r0.y*ccy[st] + r0.z*ccz[st]);
        float d2s = r1.w + cc2s[st] - 2.0f * (r1.x*cux[st] + r1.y*cuy[st] + r1.z*cuz[st]);
        float dr  = sqrtf(fmaxf(d2r, 0.0f));
        float ds_ = sqrtf(fmaxf(d2s, 0.0f));
        float dd  = dr - ds_;
        float g   = fmaxf(1.0f - dd * dd * INVS2, 0.0f);
        float scv = sv[st][j] * SCALE * g;
        sv[st][j] = scv;
        mnew[j] = fmaxf(mnew[j], scv);
      }
    }
    // --- online softmax update (half-wave reductions: xor 1,2,4,8) --------
#pragma unroll
    for (int j = 0; j < 8; ++j) {
#pragma unroll
      for (int msk = 1; msk < 16; msk <<= 1)
        mnew[j] = fmaxf(mnew[j], __shfl_xor(mnew[j], msk, 32));
      float alpha = __expf(mrun[j] - mnew[j]);
      mrun[j] = mnew[j];
      ls[j] *= alpha;
#pragma unroll
      for (int ot = 0; ot < 8; ++ot) o[ot][j] *= alpha;
    }
#pragma unroll
    for (int st = 0; st < 4; ++st) {
#pragma unroll
      for (int j = 0; j < 8; ++j) {
        float e = __expf(sv[st][j] - mrun[j]);
        ls[j] += e;
        Ps[wave][(j + roff) * kBN + st * 16 + nl] = (_Float16)e;
      }
    }
    // --- O += P @ V -------------------------------------------------------
#pragma unroll
    for (int ot = 0; ot < 8; ++ot) {
#pragma unroll
      for (int kc = 0; kc < 2; ++kc) {
        v16h a = load_frag(&Ps[wave][kc * 32], kBN);
        v16h b = load_frag(Vt + (ot * 16) * kBN + kc * 32, kBN);
        o[ot] = wmma_f16(a, b, o[ot]);
      }
    }
  }
  // --- finalize split: row stats + unnormalized partial O (f32) ----------
#pragma unroll
  for (int j = 0; j < 8; ++j) {
#pragma unroll
    for (int msk = 1; msk < 16; msk <<= 1) ls[j] += __shfl_xor(ls[j], msk, 32);
    if (nl == 0) {
      int row = mbase + j + roff;
      mstat[split * kN + row] = mrun[j];
      lstat[split * kN + row] = ls[j];
    }
  }
#pragma unroll
  for (int ot = 0; ot < 8; ++ot) {
    int col = ot * 16 + nl;
#pragma unroll
    for (int j = 0; j < 8; ++j) {
      int row = mbase + j + roff;
      Opart[((size_t)split * kN + row) * kC + col] = o[ot][j];
    }
  }
}

// ---------------------------------------------------------------------------
// merge the kSP split partials: O = sum_s Os*e^(ms-M) / sum_s ls*e^(ms-M)
// ---------------------------------------------------------------------------
__global__ void attn_merge_kernel(const float* __restrict__ Op,   // [kSP][N][C]
                                  const float* __restrict__ ms,   // [kSP][N]
                                  const float* __restrict__ lsb,  // [kSP][N]
                                  _Float16* __restrict__ O) {
  int idx = blockIdx.x * blockDim.x + threadIdx.x;
  if (idx >= kN * kC) return;
  int row = idx >> 7;
  float m[kSP], l[kSP];
  float M = -1e30f;
#pragma unroll
  for (int s = 0; s < kSP; ++s) {
    m[s] = ms[s * kN + row];
    l[s] = lsb[s * kN + row];
    M = fmaxf(M, m[s]);
  }
  float denom = 0.0f, val = 0.0f;
#pragma unroll
  for (int s = 0; s < kSP; ++s) {
    float w = __expf(m[s] - M);
    denom += l[s] * w;
    val   += Op[(size_t)s * kN * kC + idx] * w;
  }
  O[idx] = (_Float16)(val / denom);
}

// ---------------------------------------------------------------------------
// head: L2-normalize rows, 128->32->32->1 MLP, sigmoid. One row per block.
// ---------------------------------------------------------------------------
__global__ void head_kernel(const float* __restrict__ feat,
                            const float* __restrict__ c1W, const float* __restrict__ c1b,
                            const float* __restrict__ c2W, const float* __restrict__ c2b,
                            const float* __restrict__ c3W, const float* __restrict__ c3b,
                            float* __restrict__ out) {
  int row = blockIdx.x, t = threadIdx.x;
  __shared__ float sf[128], h1[32], h2[32], red[4];
  float x = feat[row * kC + t];
  float s2 = x * x;
#pragma unroll
  for (int m = 16; m >= 1; m >>= 1) s2 += __shfl_xor(s2, m, 32);
  int lane = t & 31, wv = t >> 5;
  if (lane == 0) red[wv] = s2;
  __syncthreads();
  float nrm = sqrtf(red[0] + red[1] + red[2] + red[3]);
  float nf  = x / fmaxf(nrm, 1e-12f);
  out[row * kC + t] = nf;
  sf[t] = nf;
  __syncthreads();
  if (t < 32) {
    float a = c1b[t];
    for (int k = 0; k < 128; ++k) a += sf[k] * c1W[k * 32 + t];
    h1[t] = fmaxf(a, 0.0f);
  }
  __syncthreads();
  if (t < 32) {
    float a = c2b[t];
    for (int k = 0; k < 32; ++k) a += h1[k] * c2W[k * 32 + t];
    h2[t] = fmaxf(a, 0.0f);
  }
  __syncthreads();
  if (t == 0) {
    float a = c3b[0];
    for (int k = 0; k < 32; ++k) a += h2[k] * c3W[k];
    out[kN * kC + row] = 1.0f / (1.0f + __expf(-a));
  }
}

// ---------------------------------------------------------------------------
extern "C" void kernel_launch(void* const* d_in, const int* in_sizes, int n_in,
                              void* d_out, int out_size, void* d_ws, size_t ws_size,
                              hipStream_t stream) {
  const float* refk  = (const float*)d_in[0];
  const float* srck  = (const float*)d_in[1];
  const float* cfeat = (const float*)d_in[2];
  const float* W_in  = (const float*)d_in[3];
  const float* b_in  = (const float*)d_in[4];
  const float* mlp_W = (const float*)d_in[5];
  const float* mlp_b = (const float*)d_in[6];
  const float* mlp_g = (const float*)d_in[7];
  const float* mlp_be= (const float*)d_in[8];
  const float* Wq    = (const float*)d_in[9];
  const float* bq    = (const float*)d_in[10];
  const float* Wk    = (const float*)d_in[11];
  const float* bk    = (const float*)d_in[12];
  const float* Wv    = (const float*)d_in[13];
  const float* bv    = (const float*)d_in[14];
  const float* Wo    = (const float*)d_in[15];
  const float* bo    = (const float*)d_in[16];
  const float* ln_g  = (const float*)d_in[17];
  const float* ln_b  = (const float*)d_in[18];
  const float* c1W   = (const float*)d_in[19];
  const float* c1b   = (const float*)d_in[20];
  const float* c2W   = (const float*)d_in[21];
  const float* c2b   = (const float*)d_in[22];
  const float* c3W   = (const float*)d_in[23];
  const float* c3b   = (const float*)d_in[24];
  float* out = (float*)d_out;

  // workspace carve-up (all buffers written before read every call)
  char* w = (char*)d_ws;
  auto carve = [&](size_t bytes) {
    char* p = w; w += (bytes + 255) & ~size_t(255); return p;
  };
  float*    feat  = (float*)   carve((size_t)kN * kC * 4);
  float*    tmp   = (float*)   carve((size_t)kN * kC * 4);
  _Float16* featH = (_Float16*)carve((size_t)kN * kC * 2);
  _Float16* qb    = (_Float16*)carve((size_t)kN * kC * 2);
  _Float16* kb    = (_Float16*)carve((size_t)kN * kC * 2);
  _Float16* vb    = (_Float16*)carve((size_t)kN * kC * 2);
  _Float16* ab    = (_Float16*)carve((size_t)kN * kC * 2);
  float*    Opart = (float*)   carve((size_t)kSP * kN * kC * 4);
  float*    mstat = (float*)   carve((size_t)kSP * kN * 4);
  float*    lstat = (float*)   carve((size_t)kSP * kN * 4);
  _Float16* mlpWt = (_Float16*)carve((size_t)3 * kC * kC * 2);
  _Float16* WqT   = (_Float16*)carve((size_t)3 * kC * kC * 2);
  _Float16* WkT   = (_Float16*)carve((size_t)3 * kC * kC * 2);
  _Float16* WvT   = (_Float16*)carve((size_t)3 * kC * kC * 2);
  _Float16* WoT   = (_Float16*)carve((size_t)3 * kC * kC * 2);

  const int wtot = 3 * kC * kC;
  const int wblk = (wtot + 255) / 256;
  tconv_kernel<<<wblk, 256, 0, stream>>>(mlp_W, mlpWt, wtot);
  tconv_kernel<<<wblk, 256, 0, stream>>>(Wq, WqT, wtot);
  tconv_kernel<<<wblk, 256, 0, stream>>>(Wk, WkT, wtot);
  tconv_kernel<<<wblk, 256, 0, stream>>>(Wv, WvT, wtot);
  tconv_kernel<<<wblk, 256, 0, stream>>>(Wo, WoT, wtot);

  embed_kernel<<<(kN * kC + 255) / 256, 256, 0, stream>>>(
      refk, srck, cfeat, W_in, b_in, feat, featH);

  const int gGemm = kN / 16;   // 384
  for (int l = 0; l < 3; ++l) {
    const int wo = l * kC * kC;
    const int co = l * kC;
    // UnaryBlock: linear -> LN -> LeakyReLU
    gemm_nt_kernel<0><<<gGemm, 256, 0, stream>>>(featH, mlpWt + wo, mlp_b + co, tmp);
    ln_kernel<<<kN, 128, 0, stream>>>(tmp, nullptr, mlp_g + co, mlp_be + co,
                                      feat, featH, 1);
    // q, k, v projections (f16 outputs)
    gemm_nt_kernel<1><<<gGemm, 256, 0, stream>>>(featH, WqT + wo, bq + co, qb);
    gemm_nt_kernel<1><<<gGemm, 256, 0, stream>>>(featH, WkT + wo, bk + co, kb);
    gemm_nt_kernel<1><<<gGemm, 256, 0, stream>>>(featH, WvT + wo, bv + co, vb);
    // geo-masked split-K flash attention + merge
    attn_kernel<<<dim3(kN / 64, kSP), 128, 0, stream>>>(qb, kb, vb, refk, srck,
                                                        Opart, mstat, lstat);
    attn_merge_kernel<<<(kN * kC + 255) / 256, 256, 0, stream>>>(Opart, mstat,
                                                                 lstat, ab);
    // output projection + residual + LN
    gemm_nt_kernel<0><<<gGemm, 256, 0, stream>>>(ab, WoT + wo, bo + co, tmp);
    ln_kernel<<<kN, 128, 0, stream>>>(tmp, feat, ln_g + co, ln_b + co,
                                      feat, featH, 0);
  }

  head_kernel<<<kN, 128, 0, stream>>>(feat, c1W, c1b, c2W, c2b, c3W, c3b, out);
}
